// Head_37374805410451
// MI455X (gfx1250) — matrix operand order repaired and checked by reading, not compiled, not run on hardware
//
#include <hip/hip_runtime.h>
#include <hip/hip_bf16.h>

// Sizes from the reference
constexpr int kB  = 512;
constexpr int kT  = 128;
constexpr int kE  = 768;
constexpr int kH  = 256;
constexpr int kBT = kB * kT;

typedef _Float16 v16h __attribute__((ext_vector_type(16)));
typedef _Float16 v8h  __attribute__((ext_vector_type(8)));
typedef float    v8f  __attribute__((ext_vector_type(8)));
typedef float    v4f  __attribute__((ext_vector_type(4)));

__device__ __forceinline__ v8f wmma16x16x32(v16h a, v16h b, v8f c) {
    // (neg_a, A, neg_b, B, c_mod, C, reuse_a, reuse_b)
    return __builtin_amdgcn_wmma_f32_16x16x32_f16(false, a, false, b, (short)0, c,
                                                  false, false);
}

// A-fragment (16x32 f16): lane half h holds K = {16h..16h+15} of its row.
__device__ __forceinline__ v16h load_a_frag(const _Float16* row, int h) {
    v8h lo = *(const v8h*)(row + h * 16);
    v8h hi = *(const v8h*)(row + h * 16 + 8);
    return __builtin_shufflevector(lo, hi, 0, 1, 2, 3, 4, 5, 6, 7,
                                   8, 9, 10, 11, 12, 13, 14, 15);
}

// B-fragment (32x16 f16): lane (h, c) holds column c, rows h*16 .. h*16+15.
__device__ __forceinline__ v16h load_b_frag(const _Float16* p) {
    v8h lo = *(const v8h*)(p);
    v8h hi = *(const v8h*)(p + 8);
    return __builtin_shufflevector(lo, hi, 0, 1, 2, 3, 4, 5, 6, 7,
                                   8, 9, 10, 11, 12, 13, 14, 15);
}

// ---- CDNA5 async global->LDS copy (ASYNCcnt-tracked, no VGPR transit) ----
// Wave-relative LDS byte offset = low 32 bits of the generic shared pointer
// (addrspacecast AS3->generic is {aperture_hi, lds_offset32}).
__device__ __forceinline__ void async_copy_b128(void* lds_dst, const void* gsrc) {
    asm volatile("global_load_async_to_lds_b128 %0, %1, off"
                 :
                 : "v"((uint32_t)(uintptr_t)lds_dst), "v"(gsrc)
                 : "memory");
}
__device__ __forceinline__ void wait_asynccnt0() {
    asm volatile("s_wait_asynccnt 0x0" ::: "memory");
}

// ---------------------------------------------------------------------------
// Kernel 1: QKV projection  [BT x E] (f32)  x  [E x H] (f32, per matrix)
//           -> qkv (f16)  laid out as [3][BT][H]
// Block tile: 128 rows x 64 cols, 256 threads = 8 waves.
// Double-buffered LDS staging so global f32 loads + cvt overlap the WMMAs.
// ---------------------------------------------------------------------------
constexpr int XP = 40;  // padded pitch (elems) for 128x32 X tile
constexpr int WP = 40;  // padded pitch (elems) for 64x32 transposed W tile

__device__ __forceinline__ void stage_proj_tile(const float* __restrict__ X,
                                                const float* __restrict__ Wm,
                                                int row0, int lc0, int k0, int tid,
                                                _Float16* __restrict__ Xs,
                                                _Float16* __restrict__ Ws) {
    // X tile: 128x32 f32 -> f16 (1024 float4 loads across 256 threads)
#pragma unroll
    for (int i = 0; i < 4; ++i) {
        int idx = tid + i * 256;
        int r   = idx >> 3;
        int c4  = (idx & 7) << 2;
        v4f x = *(const v4f*)(X + (size_t)(row0 + r) * kE + k0 + c4);
        Xs[r * XP + c4 + 0] = (_Float16)x.x;
        Xs[r * XP + c4 + 1] = (_Float16)x.y;
        Xs[r * XP + c4 + 2] = (_Float16)x.z;
        Xs[r * XP + c4 + 3] = (_Float16)x.w;
    }
    // W tile transposed: Ws[c][kk] = W[k0+kk][lc0+c]
#pragma unroll
    for (int i = 0; i < 8; ++i) {
        int idx = tid + i * 256;
        int kk  = idx >> 6;
        int c   = idx & 63;
        Ws[c * WP + kk] = (_Float16)Wm[(size_t)(k0 + kk) * kH + lc0 + c];
    }
}

__global__ __launch_bounds__(256)
void proj_qkv_kernel(const float* __restrict__ X,
                     const float* __restrict__ Wq,
                     const float* __restrict__ Wk,
                     const float* __restrict__ Wv,
                     _Float16* __restrict__ qkv) {
    __shared__ _Float16 Xs[2][128 * XP];
    __shared__ _Float16 Ws[2][64 * WP];

    const int tid  = threadIdx.x;
    const int wave = tid >> 5;
    const int lane = tid & 31;
    const int h    = lane >> 4;
    const int lr   = lane & 15;

    const int row0 = blockIdx.x * 128;         // token row base
    const int gc0  = blockIdx.y * 64;          // global col in [0, 3*H)
    const int m    = gc0 >> 8;                 // 0=Q, 1=K, 2=V (64 | 256)
    const int lc0  = gc0 & (kH - 1);
    const float* Wm = (m == 0) ? Wq : (m == 1) ? Wk : Wv;
    _Float16* outm = qkv + (size_t)m * kBT * kH;

    v8f acc[4];
#pragma unroll
    for (int n = 0; n < 4; ++n) acc[n] = (v8f)0.0f;

    constexpr int kTiles = kE / 32;  // 24
    stage_proj_tile(X, Wm, row0, lc0, 0, tid, Xs[0], Ws[0]);
    __syncthreads();

    const int ar = 16 * wave + lr;  // this lane's A row within the block tile
    for (int t = 0; t < kTiles; ++t) {
        const int p = t & 1;
        if (t + 1 < kTiles)
            stage_proj_tile(X, Wm, row0, lc0, (t + 1) * 32, tid, Xs[p ^ 1], Ws[p ^ 1]);

        v16h a = load_a_frag(&Xs[p][ar * XP], h);
#pragma unroll
        for (int n = 0; n < 4; ++n) {
            v16h b = load_b_frag(&Ws[p][(16 * n + lr) * WP + h * 16]);
            acc[n] = wmma16x16x32(a, b, acc[n]);
        }
        __syncthreads();
    }

    // C/D layout: VGPR i -> row (i + 8*h), col = lr
#pragma unroll
    for (int n = 0; n < 4; ++n) {
        int col = lc0 + 16 * n + lr;
#pragma unroll
        for (int i = 0; i < 8; ++i) {
            int gr = row0 + 16 * wave + i + 8 * h;
            outm[(size_t)gr * kH + col] = (_Float16)acc[n][i];
        }
    }
}

// ---------------------------------------------------------------------------
// Kernel 2: fused causal attention, one workgroup per batch.
// LDS: Qs[128x264] f16, Ks[128x264] f16, Vt[256x136] f16, Ss[128x132] f32.
// P (f16) reuses the Qs region after scores are materialized.  Total 272,384 B.
// Q/K tiles arrive via global_load_async_to_lds_b128 (ASYNCcnt path).
// ---------------------------------------------------------------------------
constexpr int QP = 264;
constexpr int KP = 264;
constexpr int VP = 136;
constexpr int SP = 132;
constexpr int PP = 136;
constexpr size_t kQBytes = (size_t)kT * QP * 2;
constexpr size_t kKBytes = (size_t)kT * KP * 2;
constexpr size_t kVBytes = (size_t)kH * VP * 2;
constexpr size_t kSBytes = (size_t)kT * SP * 4;
constexpr size_t kAttnLds = kQBytes + kKBytes + kVBytes + kSBytes;  // 272384

__global__ __launch_bounds__(256)
void attn_head_kernel(const _Float16* __restrict__ qkv,
                      float* __restrict__ out) {
    extern __shared__ char smem[];
    _Float16* Qs = (_Float16*)(smem);
    _Float16* Ks = (_Float16*)(smem + kQBytes);
    _Float16* Vt = (_Float16*)(smem + kQBytes + kKBytes);
    float*    Ss = (float*)(smem + kQBytes + kKBytes + kVBytes);
    _Float16* Ps = Qs;  // P overwrites Q after the score phase

    const int b    = blockIdx.x;
    const int tid  = threadIdx.x;
    const int wave = tid >> 5;
    const int lane = tid & 31;
    const int h    = lane >> 4;
    const int lr   = lane & 15;

    const _Float16* qg = qkv + (size_t)b * kT * kH;
    const _Float16* kg = qkv + (size_t)kBT * kH + (size_t)b * kT * kH;
    const _Float16* vg = qkv + (size_t)2 * kBT * kH + (size_t)b * kT * kH;

    // ---- Q, K: async global->LDS (layout-preserving 16B chunks) ----
#pragma unroll
    for (int i = 0; i < 16; ++i) {
        int idx = tid + i * 256;          // 4096 chunks of 8 f16
        int r   = idx >> 5;
        int c8  = (idx & 31) << 3;
        async_copy_b128(&Qs[r * QP + c8], qg + r * kH + c8);
        async_copy_b128(&Ks[r * KP + c8], kg + r * kH + c8);
    }
    // ---- V: VGPR transit (needs transpose into Vt) ----
#pragma unroll
    for (int i = 0; i < 16; ++i) {
        int idx = tid + i * 256;
        int r   = idx >> 5;
        int c8  = (idx & 31) << 3;
        v8h v = *(const v8h*)(vg + r * kH + c8);
#pragma unroll
        for (int j = 0; j < 8; ++j) Vt[(c8 + j) * VP + r] = v[j];
    }
    wait_asynccnt0();
    __syncthreads();

    // ---- Scores: S = Q K^T * E^-0.5 ; wave w owns rows 16w..16w+15 ----
    v8f sacc[8];
#pragma unroll
    for (int n = 0; n < 8; ++n) sacc[n] = (v8f)0.0f;
    const int ar = 16 * wave + lr;
    for (int k0 = 0; k0 < kH; k0 += 32) {
        v16h a = load_a_frag(&Qs[ar * QP + k0], h);
#pragma unroll
        for (int n = 0; n < 8; ++n) {
            // B[kk][s] = K[16n+lr][k0+kk] -> contiguous in row-major K
            v16h bf = load_b_frag(&Ks[(16 * n + lr) * KP + k0 + h * 16]);
            sacc[n] = wmma16x16x32(a, bf, sacc[n]);
        }
    }
    const float scale = rsqrtf((float)kE);  // reference scales by embed^-0.5
#pragma unroll
    for (int n = 0; n < 8; ++n)
#pragma unroll
        for (int i = 0; i < 8; ++i)
            Ss[(16 * wave + i + 8 * h) * SP + 16 * n + lr] = sacc[n][i] * scale;
    __syncthreads();

    // ---- Row-wise causal softmax; write P (f16) into the old Q region ----
    if (tid < kT) {
        const int r = tid;
        float mx = -3.0e38f;
        for (int c = 0; c <= r; ++c) mx = fmaxf(mx, Ss[r * SP + c]);
        float sum = 0.0f;
        for (int c = 0; c <= r; ++c) sum += __expf(Ss[r * SP + c] - mx);
        const float inv = 1.0f / sum;
        for (int c = 0; c < kT; ++c) {
            float p = (c <= r) ? __expf(Ss[r * SP + c] - mx) * inv : 0.0f;
            Ps[r * PP + c] = (_Float16)p;
        }
    }
    __syncthreads();

    // ---- O = P @ V ; wave w owns rows 16w..16w+15, all 16 col tiles ----
    for (int n = 0; n < 16; ++n) {
        v8f acc = (v8f)0.0f;
#pragma unroll
        for (int k0 = 0; k0 < kT; k0 += 32) {
            v16h a  = load_a_frag(&Ps[ar * PP + k0], h);
            v16h bf = load_b_frag(&Vt[(16 * n + lr) * VP + k0 + h * 16]);
            acc = wmma16x16x32(a, bf, acc);
        }
#pragma unroll
        for (int i = 0; i < 8; ++i) {
            int gr = 16 * wave + i + 8 * h;
            out[((size_t)b * kT + gr) * kH + 16 * n + lr] = acc[i];
        }
    }
}

// ---------------------------------------------------------------------------
// Host entry
// ---------------------------------------------------------------------------
extern "C" void kernel_launch(void* const* d_in, const int* in_sizes, int n_in,
                              void* d_out, int out_size, void* d_ws, size_t ws_size,
                              hipStream_t stream) {
    // setup_inputs() dict order: res_stream, Wk, Wq, Wv
    const float* X  = (const float*)d_in[0];
    const float* Wk = (const float*)d_in[1];
    const float* Wq = (const float*)d_in[2];
    const float* Wv = (const float*)d_in[3];
    float* out = (float*)d_out;

    // Workspace: QKV in f16, [3][BT][H]  (~96 MB)
    _Float16* qkv = (_Float16*)d_ws;

    // Raise the dynamic-LDS cap for the 272 KB attention kernel (CDNA5: 320 KB/WGP).
    (void)hipFuncSetAttribute((const void*)attn_head_kernel,
                              hipFuncAttributeMaxDynamicSharedMemorySize,
                              (int)kAttnLds);

    dim3 pgrid(kBT / 128, (3 * kH) / 64);
    proj_qkv_kernel<<<pgrid, 256, 0, stream>>>(X, Wq, Wk, Wv, qkv);

    attn_head_kernel<<<dim3(kB), 256, kAttnLds, stream>>>(qkv, out);
}
